// AffinityNodeLayer_Cora_91010357002494
// MI455X (gfx1250) — compile-verified, matrix operand order checked by reading, not run on hardware
//
#include <hip/hip_runtime.h>
#include <stdint.h>

#define N_NODES  4096
#define IN_FEAT  1433
#define K_PAD    1440          // 45 * 32
#define N_HEADS  8
#define N_HIDDEN 64
#define HID      (N_HEADS * N_HIDDEN)   // 512
#define NTILE    (N_NODES / 16)         // 256
#define NSUP     (N_NODES / 64)         // 64 super-tiles per dim
#define SUP_BYTES (64 * HID * 2)        // 64 rows of g in bytes = 64 KB

typedef __attribute__((ext_vector_type(16))) __bf16 bf16x16;
typedef __attribute__((ext_vector_type(8)))  float  f32x8;
typedef __attribute__((ext_vector_type(2)))  float  f32x2;

union FragU { uint4 q[2]; bf16x16 v; };

// f32 -> bf16 round-to-nearest-even
__device__ __forceinline__ uint16_t f2bf(float x) {
  union { float f; uint32_t u; } c; c.f = x;
  uint32_t u = c.u;
  uint32_t r = (u + 0x7FFFu + ((u >> 16) & 1u)) >> 16;
  return (uint16_t)r;
}

// 16x32 bf16 row-fragment (A-layout, wave32):
//   lanes 0-15  : M = lane,    K {0..7} (q0) and {16..23} (q1)
//   lanes 16-31 : M = lane-16, K {8..15}     and {24..31}
__device__ __forceinline__ bf16x16 load_frag(const uint16_t* __restrict__ base,
                                             int ld, int row0, int col0, int lane) {
  const int row  = row0 + (lane & 15);
  const int half = lane >> 4;
  const uint16_t* p = base + (size_t)row * ld + col0 + 8 * half;
  FragU f;
  f.q[0] = *(const uint4*)(p);
  f.q[1] = *(const uint4*)(p + 16);
  return f.v;
}

// --------------------------- packed f32 FMA (VOP3P) ------------------------
// Plain packed fma (no op_sel: gfx1250 VOP3P f32 repurposes OPSEL as reuse
// hints, so only the modifier-free form is encodable).
__device__ __forceinline__ f32x2 pk_fma_vv(f32x2 a, f32x2 b, f32x2 c) {
  f32x2 d;
  asm("v_pk_fma_f32 %0, %1, %2, %3"
      : "=v"(d) : "v"(a), "v"(b), "v"(c));
  return d;
}

// --------------------------- async global -> LDS ---------------------------
// CDNA5 GLOBAL_LOAD_ASYNC_TO_LDS_B128 (VGLOBAL op 98): per-lane 16B mem->LDS,
// tracked by ASYNCcnt. vdst = VGPR holding the LDS byte address,
// vaddr = 64-bit global address, saddr = off (GV mode).
__device__ __forceinline__ void async_ld_b128(const uint16_t* gsrc, uint32_t lds_byte_off) {
  const uint64_t ga = (uint64_t)(uintptr_t)gsrc;
  asm volatile("global_load_async_to_lds_b128 %0, %1, off"
               :: "v"(lds_byte_off), "v"(ga) : "memory");
}

__device__ __forceinline__ void wait_async0() {
#if defined(__has_builtin) && __has_builtin(__builtin_amdgcn_s_wait_asynccnt)
  __builtin_amdgcn_s_wait_asynccnt(0);
#else
  asm volatile("s_wait_asynccnt 0" ::: "memory");
#endif
}

__device__ __forceinline__ uint32_t lds_off32(const void* p) {
  // generic LDS pointer: addr[31:0] is the LDS offset (ISA 10.2 aperture rules)
  return (uint32_t)(uintptr_t)p;
}

// ---------------------------------------------------------------------------
// Kernel 0: convert vertex [4096,1433] and W_vert [512,1433] to bf16, padded.
// ---------------------------------------------------------------------------
__global__ void cvt_pad_kernel(const float* __restrict__ vertex,
                               const float* __restrict__ Wvert,
                               uint16_t* __restrict__ Vb,
                               uint16_t* __restrict__ Wb) {
  const int idx = blockIdx.x * blockDim.x + threadIdx.x;
  const int NV = N_NODES * K_PAD;
  const int NW = HID * K_PAD;
  if (idx < NV) {
    const int row = idx / K_PAD, col = idx - row * K_PAD;
    const float v = (col < IN_FEAT) ? vertex[(size_t)row * IN_FEAT + col] : 0.0f;
    Vb[idx] = f2bf(v);
  } else if (idx < NV + NW) {
    const int j = idx - NV;
    const int row = j / K_PAD, col = j - row * K_PAD;
    const float v = (col < IN_FEAT) ? Wvert[(size_t)row * IN_FEAT + col] : 0.0f;
    Wb[j] = f2bf(v);
  }
}

// ---------------------------------------------------------------------------
// Kernel 1: g = vertex @ W_vert.T  ->  Gb bf16 [4096, 512]
// ---------------------------------------------------------------------------
__global__ void __launch_bounds__(256)
gemm_g_kernel(const uint16_t* __restrict__ Vb,
              const uint16_t* __restrict__ Wb,
              uint16_t* __restrict__ Gb) {
  const int lane = threadIdx.x & 31;
  const int wave = blockIdx.x * (blockDim.x >> 5) + (threadIdx.x >> 5);
  const int tn = wave >> 5;
  const int tc = wave & 31;

  f32x8 acc = {};
  for (int k = 0; k < K_PAD; k += 32) {
    bf16x16 a = load_frag(Vb, K_PAD, tn * 16, k, lane);
    bf16x16 b = load_frag(Wb, K_PAD, tc * 16, k, lane);
    acc = __builtin_amdgcn_wmma_f32_16x16x32_bf16(false, a, false, b,
                                                  (short)0, acc, false, false);
  }

  const int col   = tc * 16 + (lane & 15);
  const int rbase = tn * 16 + 8 * (lane >> 4);
#pragma unroll
  for (int r = 0; r < 8; ++r)
    Gb[(size_t)(rbase + r) * HID + col] = f2bf(acc[r]);
}

// ---------------------------------------------------------------------------
// Epilogue for one 16x16 tile: rows in adjacent pairs; head-mix via
// elementwise fma on float2, lrelu-combine via v_pk_fma_f32.
//   out = sum_g lrelu(sum_h gram_h * W_attn[g,h]),  lrelu(x) = 0.6x + 0.4|x|
// Store tile at (ni,mi) and optionally its mirror at (mi,ni).
// ---------------------------------------------------------------------------
__device__ __forceinline__ void epilogue_store(const f32x8* acc, const float (*w)[8],
                                               int ni, int mi, bool mirror,
                                               int lane, float* __restrict__ out) {
  const int half = lane >> 4;
  const int cn   = lane & 15;
  const f32x2 c06 = {0.6f, 0.6f};
  const f32x2 c04 = {0.4f, 0.4f};
  f32x2 val2[4];
#pragma unroll
  for (int pr = 0; pr < 4; ++pr) {
    f32x2 gr[8];
#pragma unroll
    for (int h = 0; h < 8; ++h) {
      f32x2 t; t.x = acc[h][2 * pr]; t.y = acc[h][2 * pr + 1];
      gr[h] = t;
    }
    f32x2 s = {0.0f, 0.0f};
#pragma unroll
    for (int g = 0; g < 8; ++g) {
      f32x2 m = {0.0f, 0.0f};
#pragma unroll
      for (int h = 0; h < 8; ++h) {
        const f32x2 ws = {w[g][h], w[g][h]};
        m = __builtin_elementwise_fma(ws, gr[h], m);
      }
      const f32x2 am = __builtin_elementwise_abs(m);
      s = pk_fma_vv(c06, m, s);
      s = pk_fma_vv(c04, am, s);
    }
    val2[pr] = s;
  }

  {
    const int col   = mi * 16 + cn;
    const int rbase = ni * 16 + 8 * half;
#pragma unroll
    for (int r = 0; r < 8; ++r)
      out[(size_t)(rbase + r) * N_NODES + col] = val2[r >> 1][r & 1];
  }
  if (mirror) {
    const int row = mi * 16 + cn;
    float4* dst = (float4*)(out + (size_t)row * N_NODES + ni * 16 + 8 * half);
    dst[0] = make_float4(val2[0].x, val2[0].y, val2[1].x, val2[1].y);
    dst[1] = make_float4(val2[2].x, val2[2].y, val2[3].x, val2[3].y);
  }
}

// ---------------------------------------------------------------------------
// Kernel 2: one block = one 64x64 super-tile pair (NI <= MI).
//  - async-copy 64 rows of g for NI and MI into LDS (2 x 64 KB)
//  - 8 waves x 2 tiles (shared A fragment), 8 heads x 4 WMMA per wave
//  - diagonal super-tiles computed fully (no mirror); off-diagonal mirrored
// ---------------------------------------------------------------------------
__global__ void __launch_bounds__(256)
affinity_kernel2(const uint16_t* __restrict__ Gb,
                 const float* __restrict__ Wattn,
                 float* __restrict__ out) {
  extern __shared__ uint16_t smem[];           // [2][64][512] bf16 = 128 KB
  uint16_t* Sn = smem;
  uint16_t* Sm = smem + 64 * HID;

  const int tid  = threadIdx.x;
  const int lane = tid & 31;
  const int wave = tid >> 5;

  // decode upper-triangular super-pair (NI <= MI), C(i) = i*NSUP - i*(i-1)/2
  const int p = blockIdx.x;
  int i = (int)(64.5f - sqrtf(64.5f * 64.5f - 2.0f * (float)p));
  if (i < 0) i = 0;
  if (i > NSUP - 1) i = NSUP - 1;
  while ((i + 1) * NSUP - (((i + 1) * i) >> 1) <= p) ++i;
  while (i * NSUP - ((i * (i - 1)) >> 1) > p) --i;
  const int NI = i;
  const int MI = NI + (p - (i * NSUP - ((i * (i - 1)) >> 1)));

  // ---- stage 64+64 rows of g into LDS via async loads (16B per lane) ----
  {
    const uint16_t* gn = Gb + (size_t)NI * 64 * HID;
    const uint16_t* gm = Gb + (size_t)MI * 64 * HID;
    const uint32_t sn_off = lds_off32(Sn);
    const uint32_t sm_off = lds_off32(Sm);
#pragma unroll
    for (int it = 0; it < 16; ++it) {          // 4096 16B-chunks per region / 256 thr
      const int chunk = it * 256 + tid;        // chunk * 8 uint16 = 16 bytes
      async_ld_b128(gn + chunk * 8, sn_off + chunk * 16);
      async_ld_b128(gm + chunk * 8, sm_off + chunk * 16);
    }
    wait_async0();
    __syncthreads();
  }

  // W_attn (8x8) — uniform scalar loads
  float w[8][8];
#pragma unroll
  for (int g = 0; g < 8; ++g)
#pragma unroll
    for (int h = 0; h < 8; ++h)
      w[g][h] = Wattn[g * 8 + h];

  // wave -> two tiles sharing A: local a = wave>>1, b0 = 2*(wave&1), b1 = b0+1
  const int a  = wave >> 1;
  const int b0 = (wave & 1) * 2;
  const int b1 = b0 + 1;

  f32x8 acc0[N_HEADS] = {};
  f32x8 acc1[N_HEADS] = {};
#pragma unroll
  for (int h = 0; h < N_HEADS; ++h) {
    const int c = h * N_HIDDEN;
    bf16x16 A0  = load_frag(Sn, HID, a * 16,  c,      lane);
    bf16x16 A1  = load_frag(Sn, HID, a * 16,  c + 32, lane);
    bf16x16 B00 = load_frag(Sm, HID, b0 * 16, c,      lane);
    bf16x16 B01 = load_frag(Sm, HID, b0 * 16, c + 32, lane);
    bf16x16 B10 = load_frag(Sm, HID, b1 * 16, c,      lane);
    bf16x16 B11 = load_frag(Sm, HID, b1 * 16, c + 32, lane);
    acc0[h] = __builtin_amdgcn_wmma_f32_16x16x32_bf16(false, A0, false, B00,
                                                      (short)0, acc0[h], false, false);
    acc0[h] = __builtin_amdgcn_wmma_f32_16x16x32_bf16(false, A1, false, B01,
                                                      (short)0, acc0[h], false, false);
    acc1[h] = __builtin_amdgcn_wmma_f32_16x16x32_bf16(false, A0, false, B10,
                                                      (short)0, acc1[h], false, false);
    acc1[h] = __builtin_amdgcn_wmma_f32_16x16x32_bf16(false, A1, false, B11,
                                                      (short)0, acc1[h], false, false);
  }

  const bool mirror = (NI != MI);
  epilogue_store(acc0, w, NI * 4 + a, MI * 4 + b0, mirror, lane, out);
  epilogue_store(acc1, w, NI * 4 + a, MI * 4 + b1, mirror, lane, out);
}

// ---------------------------------------------------------------------------
extern "C" void kernel_launch(void* const* d_in, const int* in_sizes, int n_in,
                              void* d_out, int out_size, void* d_ws, size_t ws_size,
                              hipStream_t stream) {
  const float* vertex = (const float*)d_in[0];
  const float* Wvert  = (const float*)d_in[1];
  const float* Wattn  = (const float*)d_in[2];
  float* out = (float*)d_out;

  char* ws = (char*)d_ws;
  uint16_t* Vb = (uint16_t*)(ws);
  uint16_t* Wb = (uint16_t*)(ws + (size_t)N_NODES * K_PAD * 2);
  uint16_t* Gb = (uint16_t*)(ws + (size_t)N_NODES * K_PAD * 2
                                + (size_t)HID * K_PAD * 2);

  {
    const int total = N_NODES * K_PAD + HID * K_PAD;
    cvt_pad_kernel<<<(total + 255) / 256, 256, 0, stream>>>(vertex, Wvert, Vb, Wb);
  }
  {
    const int waves = (N_NODES / 16) * (HID / 16);   // 8192 tiles, 8 waves/block
    gemm_g_kernel<<<waves / 8, 256, 0, stream>>>(Vb, Wb, Gb);
  }
  {
    const int pairs = NSUP * (NSUP + 1) / 2;          // 2080 super-tile pairs
    const size_t lds_bytes = 2 * SUP_BYTES;           // 128 KB
    (void)hipFuncSetAttribute((const void*)affinity_kernel2,
                              hipFuncAttributeMaxDynamicSharedMemorySize,
                              (int)lds_bytes);
    affinity_kernel2<<<pairs, 256, lds_bytes, stream>>>(Gb, Wattn, out);
  }
}
